// Generator_44547400794492
// MI455X (gfx1250) — compile-verified, matrix operand order checked by reading, not run on hardware
//
#include <hip/hip_runtime.h>

// ---------------------------------------------------------------------------
// Generator (MLP -> output-feedback LSTM) for MI455X / gfx1250, wave32 WMMA.
//   B=8192, latent=256, hidden=512, out=512, T=16
//   t>=1: input==h  =>  gates = h @ (Wih+Whh)^T + bias   (halves LSTM FLOPs)
// Per workgroup: 32 batch rows, 8 waves; h in LDS ping-pong (XOR-swizzled,
// bank-conflict-free A fragments), c in VGPRs, weights pre-packed into WMMA
// fragment order for fully-coalesced global_load_b128 streams from L2.
// ---------------------------------------------------------------------------

typedef __attribute__((ext_vector_type(8)))  _Float16 v8h;
typedef __attribute__((ext_vector_type(16))) _Float16 v16h;
typedef __attribute__((ext_vector_type(8)))  float    v8f;

#define WMMA(a, b, c) \
  __builtin_amdgcn_wmma_f32_16x16x32_f16(false, (a), false, (b), (short)0, (c), false, false)

__device__ __forceinline__ v16h cat8(v8h c0, v8h c1) {
  return __builtin_shufflevector(c0, c1, 0,1,2,3,4,5,6,7,8,9,10,11,12,13,14,15);
}

// ---- packed B fragments: [frag][2 chunks][32 lanes][8 f16], 512 f16 each ----
// frag = n_tile*KT + kt. Chunk loads are lane-contiguous => fully coalesced.
__device__ __forceinline__ v16h load_bfrag(const _Float16* __restrict__ Wpk,
                                           int frag, int lane) {
  const _Float16* p = Wpk + ((size_t)frag << 9) + (lane << 3);
  v8h c0 = *(const v8h*)(p);
  v8h c1 = *(const v8h*)(p + 256);
  return cat8(c0, c1);
}

// ---- LDS activation tile: 32 rows x 512 f16, 16B-chunk XOR swizzle --------
// phys_chunk = (j>>3) ^ (m&15): rows of an A fragment hit 16 distinct 4-bank
// groups => minimum-phase ds_load_b128 (no 16-way conflicts), still 64KB total.
__device__ __forceinline__ int lds_off(int m, int cj) {
  return m * 512 + (((cj) ^ (m & 15)) << 3);
}
__device__ __forceinline__ v16h load_afrag(const _Float16* X, int m, int kb) {
  const int cj = kb >> 3;
  v8h c0 = *(const v8h*)(X + lds_off(m, cj));
  v8h c1 = *(const v8h*)(X + lds_off(m, cj + 2));
  return cat8(c0, c1);
}
__device__ __forceinline__ void lds_store_h(_Float16* X, int m, int j, _Float16 v) {
  X[m * 512 + (((j >> 3) ^ (m & 15)) << 3) + (j & 7)] = v;
}

__device__ __forceinline__ float fast_sigmoid(float x) { return 1.0f / (1.0f + __expf(-x)); }
__device__ __forceinline__ float fast_tanh(float x)    { return 1.0f - 2.0f / (1.0f + __expf(2.0f * x)); }

// ---------------------------------------------------------------------------
// Prep: f32->f16, fuse Wsum=Wih+Whh and biases, pack into fragment order.
// dst element d: i=d&7, lane=(d>>3)&31, chunk=(d>>8)&1, frag=d>>9,
//   kt=frag&(KT-1), nt=frag>>ktbits, n=nt*16+(lane&15),
//   k=kt*32+chunk*16+(lane>>4)*8+i
// ---------------------------------------------------------------------------
__device__ __forceinline__ void unpack_idx(int d, int ktbits, int& n, int& k) {
  const int i    = d & 7;
  const int lane = (d >> 3) & 31;
  const int ch   = (d >> 8) & 1;
  const int frag = d >> 9;
  const int kt   = frag & ((1 << ktbits) - 1);
  const int nt   = frag >> ktbits;
  n = nt * 16 + (lane & 15);
  k = kt * 32 + ch * 16 + (lane >> 4) * 8 + i;
}

__global__ void prep_kernel(const float* __restrict__ W1, const float* __restrict__ b1,
                            const float* __restrict__ W2, const float* __restrict__ W3,
                            const float* __restrict__ Wih, const float* __restrict__ Whh,
                            const float* __restrict__ bih, const float* __restrict__ bhh,
                            const int* __restrict__ slen,
                            _Float16* __restrict__ W1h, _Float16* __restrict__ W2h,
                            _Float16* __restrict__ W3h, _Float16* __restrict__ Wihh,
                            _Float16* __restrict__ Wsumh,
                            float* __restrict__ b1e, float* __restrict__ lbias) {
  const int stride = gridDim.x * blockDim.x;
  int n, k;
  for (int idx = blockIdx.x * blockDim.x + threadIdx.x; idx < 2048 * 512; idx += stride) {
    unpack_idx(idx, 4, n, k);                               // KT=16 (K=512)
    const int src = n * 512 + k;
    const float wi = Wih[src];
    Wihh[idx]  = (_Float16)wi;
    Wsumh[idx] = (_Float16)(wi + Whh[src]);                 // input-feedback sum
    if (idx < 512 * 512) {                                  // N=512, K=512
      W2h[idx] = (_Float16)W2[src];
      W3h[idx] = (_Float16)W3[src];
    }
    if (idx < 512 * 256) {                                  // N=512, K=256 (KT=8)
      unpack_idx(idx, 3, n, k);
      W1h[idx] = (_Float16)W1[n * 273 + 17 + k];            // noise block of W1
    }
    if (idx < 2048) lbias[idx] = bih[idx] + bhh[idx];
    if (idx < 512)  b1e[idx] = b1[idx] + W1[idx * 273 + (20 - *slen)];  // one-hot col
  }
}

// ---------------------------------------------------------------------------
// One dense layer on the 32-row tile: Y = act(X @ Wt^T + bias), cols per wave.
// ---------------------------------------------------------------------------
__device__ __forceinline__ void mlp_layer(const _Float16* X,
                                          const _Float16* __restrict__ Wpk, int KT,
                                          const float* __restrict__ bias,
                                          _Float16* Y, int act,
                                          int wj, int lane, int l15, int hs) {
  v8f zero = {};
#pragma unroll 1
  for (int ct = 0; ct < 4; ++ct) {
    const int n = wj * 64 + ct * 16 + l15;
    const int fbase = (wj * 4 + ct) * KT;
    v8f acc0 = zero, acc1 = zero;
#pragma unroll 1
    for (int kt = 0; kt < KT; ++kt) {
      const int kb = kt * 32 + hs * 8;
      v16h b  = load_bfrag(Wpk, fbase + kt, lane);
      v16h a0 = load_afrag(X, l15, kb);
      v16h a1 = load_afrag(X, l15 + 16, kb);
      acc0 = WMMA(a0, b, acc0);
      acc1 = WMMA(a1, b, acc1);
    }
    const float bv = bias[n];
#pragma unroll
    for (int mt = 0; mt < 2; ++mt) {
      v8f a = mt ? acc1 : acc0;
#pragma unroll
      for (int e = 0; e < 8; ++e) {
        float v = a[e] + bv;
        v = act ? fast_sigmoid(v) : fmaxf(v, 0.0f);
        lds_store_h(Y, mt * 16 + hs * 8 + e, n, (_Float16)v);
      }
    }
  }
}

// ---------------------------------------------------------------------------
// Main kernel: grid = 256 blocks x 256 threads; block b owns rows [32b, 32b+32).
// ---------------------------------------------------------------------------
__global__ __launch_bounds__(256) void gen_kernel(
    const float* __restrict__ x,
    const float* __restrict__ b2, const float* __restrict__ b3,
    const _Float16* __restrict__ W1h, const _Float16* __restrict__ W2h,
    const _Float16* __restrict__ W3h, const _Float16* __restrict__ Wihh,
    const _Float16* __restrict__ Wsumh,
    const float* __restrict__ b1e, const float* __restrict__ lbias,
    float* __restrict__ out) {
  __shared__ _Float16 hbuf[2][32 * 512];                    // 64 KB ping-pong tiles

  const int tid  = threadIdx.x;
  const int wj   = tid >> 5;                                // wave id (wave32)
  const int lane = tid & 31;
  const int l15  = lane & 15;
  const int hs   = lane >> 4;
  const int row0 = blockIdx.x * 32;

  // ---- stage x tile as f16 (swizzled) into hbuf[1], cols 0..255 ----
  _Float16* zb = &hbuf[1][0];
  for (int idx = tid; idx < 32 * 256; idx += 256) {
    int r = idx >> 8, c = idx & 255;
    lds_store_h(zb, r, c, (_Float16)x[(size_t)(row0 + r) * 256 + c]);
  }
  __syncthreads();

  // ---- MLP ----
  mlp_layer(zb,          W1h,  8, b1e, &hbuf[0][0], 0, wj, lane, l15, hs);  // relu
  __syncthreads();
  mlp_layer(&hbuf[0][0], W2h, 16, b2,  &hbuf[1][0], 0, wj, lane, l15, hs);  // relu
  __syncthreads();
  mlp_layer(&hbuf[1][0], W3h, 16, b3,  &hbuf[0][0], 1, wj, lane, l15, hs);  // sigmoid -> x0
  __syncthreads();

  // ---- LSTM: 16 steps, c in registers, h ping-pong in LDS ----
  v8f zero = {};
  v8f cst[4][2];
#pragma unroll
  for (int ct = 0; ct < 4; ++ct)
#pragma unroll
    for (int mt = 0; mt < 2; ++mt) cst[ct][mt] = zero;

  int cur = 0;
  for (int t = 0; t < 16; ++t) {
    const _Float16* Wt   = (t == 0) ? Wihh : Wsumh;         // h0==0 -> Wih only at t=0
    const _Float16* hin  = &hbuf[cur][0];
    _Float16*       hout = &hbuf[cur ^ 1][0];

#pragma unroll
    for (int ct = 0; ct < 4; ++ct) {
      const int jc = wj * 64 + ct * 16 + l15;               // output column (0..511)
      const int nt = wj * 4 + ct;                           // column tile within a gate
      v8f aI0 = zero, aI1 = zero, aF0 = zero, aF1 = zero;
      v8f aG0 = zero, aG1 = zero, aO0 = zero, aO1 = zero;
#pragma unroll 1
      for (int kt = 0; kt < 16; ++kt) {
        const int kb = kt * 32 + hs * 8;
        v16h bi = load_bfrag(Wt, (nt      ) * 16 + kt, lane);   // gate i: tiles   0..31
        v16h bf = load_bfrag(Wt, (nt +  32) * 16 + kt, lane);   // gate f: tiles  32..63
        v16h bg = load_bfrag(Wt, (nt +  64) * 16 + kt, lane);   // gate g: tiles  64..95
        v16h bo = load_bfrag(Wt, (nt +  96) * 16 + kt, lane);   // gate o: tiles  96..127
        v16h a0 = load_afrag(hin, l15, kb);
        v16h a1 = load_afrag(hin, l15 + 16, kb);
        aI0 = WMMA(a0, bi, aI0); aI1 = WMMA(a1, bi, aI1);
        aF0 = WMMA(a0, bf, aF0); aF1 = WMMA(a1, bf, aF1);
        aG0 = WMMA(a0, bg, aG0); aG1 = WMMA(a1, bg, aG1);
        aO0 = WMMA(a0, bo, aO0); aO1 = WMMA(a1, bo, aO1);
      }
      const float Bi = lbias[jc], Bf = lbias[jc + 512];
      const float Bg = lbias[jc + 1024], Bo = lbias[jc + 1536];
#pragma unroll
      for (int mt = 0; mt < 2; ++mt) {
        v8f vI = mt ? aI1 : aI0, vF = mt ? aF1 : aF0;
        v8f vG = mt ? aG1 : aG0, vO = mt ? aO1 : aO0;
#pragma unroll
        for (int e = 0; e < 8; ++e) {
          const int m = mt * 16 + hs * 8 + e;               // row within 32-row tile
          float iv = fast_sigmoid(vI[e] + Bi);
          float fv = fast_sigmoid(vF[e] + Bf);
          float gv = fast_tanh(vG[e] + Bg);
          float ov = fast_sigmoid(vO[e] + Bo);
          float c  = fv * cst[ct][mt][e] + iv * gv;
          cst[ct][mt][e] = c;
          float hv = ov * fast_tanh(c);
          lds_store_h(hout, m, jc, (_Float16)hv);           // feedback input for t+1
          out[((size_t)(row0 + m) * 16 + t) * 512 + jc] = hv;
        }
      }
    }
    __syncthreads();
    cur ^= 1;
  }
}

// ---------------------------------------------------------------------------
// Launch
// ---------------------------------------------------------------------------
extern "C" void kernel_launch(void* const* d_in, const int* in_sizes, int n_in,
                              void* d_out, int out_size, void* d_ws, size_t ws_size,
                              hipStream_t stream) {
  const float* x   = (const float*)d_in[0];
  const float* W1  = (const float*)d_in[1];
  const float* b1  = (const float*)d_in[2];
  const float* W2  = (const float*)d_in[3];
  const float* b2  = (const float*)d_in[4];
  const float* W3  = (const float*)d_in[5];
  const float* b3  = (const float*)d_in[6];
  const float* Wih = (const float*)d_in[7];
  const float* Whh = (const float*)d_in[8];
  const float* bih = (const float*)d_in[9];
  const float* bhh = (const float*)d_in[10];
  const int*  slen = (const int*)d_in[11];

  char* ws = (char*)d_ws;
  _Float16* W1h   = (_Float16*)(ws + 0);        // 512*256*2  = 262144
  _Float16* W2h   = (_Float16*)(ws + 262144);   // 512*512*2  = 524288
  _Float16* W3h   = (_Float16*)(ws + 786432);   // 512*512*2  = 524288
  _Float16* Wihh  = (_Float16*)(ws + 1310720);  // 2048*512*2 = 2097152
  _Float16* Wsumh = (_Float16*)(ws + 3407872);  // 2048*512*2 = 2097152
  float*    b1e   = (float*)(ws + 5505024);     // 512*4
  float*    lbias = (float*)(ws + 5507072);     // 2048*4

  prep_kernel<<<1024, 256, 0, stream>>>(W1, b1, W2, W3, Wih, Whh, bih, bhh, slen,
                                        W1h, W2h, W3h, Wihh, Wsumh, b1e, lbias);
  gen_kernel<<<256, 256, 0, stream>>>(x, b2, b3, W1h, W2h, W3h, Wihh, Wsumh,
                                      b1e, lbias, (float*)d_out);
}
